// CrossAttentionNoGate_51694226374802
// MI455X (gfx1250) — matrix operand order repaired and checked by reading, not compiled, not run on hardware
//
#include <hip/hip_runtime.h>

// ---------- types ----------
typedef __attribute__((ext_vector_type(16))) _Float16 v16h;
typedef __attribute__((ext_vector_type(8)))  _Float16 v8h;
typedef __attribute__((ext_vector_type(8)))  float    v8f;

#define WMMA_F16(a, b, c) \
  __builtin_amdgcn_wmma_f32_16x16x32_f16(false, (a), false, (b), (short)0, (c), false, false)

// Problem constants
static constexpr int S_  = 64;
static constexpr int QL  = 512;
static constexpr int KVL = 512;
static constexpr int CQ  = 256;
static constexpr int H_  = 8;
static constexpr int D_  = 32;

// ---------- CDNA5 async global->LDS copy (16B per lane, ASYNCcnt-tracked) ----------
__device__ inline void async_copy_b128(const _Float16* gsrc, _Float16* ldst) {
  // Generic shared pointer: low 32 bits are the LDS byte offset.
  unsigned lds_addr = (unsigned)(unsigned long long)(const void*)ldst;
  asm volatile("global_load_async_to_lds_b128 %0, %1, off"
               :: "v"(lds_addr), "v"(gsrc)
               : "memory");
}
__device__ inline void async_wait0() {
  asm volatile("s_wait_asynccnt 0" ::: "memory");
}

// ---------- WMMA fragment loaders (CDNA5 16x16x32 f16 layouts) ----------
// A (16x32, MxK) from row-major storage (row stride in halves, multiple of 8).
// lanes 0-15: row M=lane, K in {0..7, 16..23}; lanes 16-31: row M=lane-16, K in {8..15, 24..31}
__device__ inline v16h load_frag_a(const _Float16* base, int stride) {
  int lane = threadIdx.x & 31;
  int hb = lane >> 4, m = lane & 15;
  const _Float16* p = base + (size_t)m * stride;
  v8h lo = *(const v8h*)(p + hb * 8);
  v8h hi = *(const v8h*)(p + 16 + hb * 8);
  return __builtin_shufflevector(lo, hi, 0,1,2,3,4,5,6,7,8,9,10,11,12,13,14,15);
}

// B (32x16, KxN) from TRANSPOSED storage bT[n][k] (row stride in halves, multiple of 8).
// lanes 0-15: col N=lane, K in {0..7, 16..23}; lanes 16-31: col N=lane-16, K in {8..15, 24..31}
__device__ inline v16h load_frag_b(const _Float16* baseT, int stride) {
  int lane = threadIdx.x & 31;
  int hb = lane >> 4, n = lane & 15;
  const _Float16* p = baseT + (size_t)n * stride;
  v8h lo = *(const v8h*)(p + hb * 8);
  v8h hi = *(const v8h*)(p + 16 + hb * 8);
  return __builtin_shufflevector(lo, hi, 0,1,2,3,4,5,6,7,8,9,10,11,12,13,14,15);
}

// ---------- projection GEMM: C[M,N] = A[M,256] * W[256,N] ----------
// MODE 0: A=f32 input_q,  store q  f16 [s][h][q][d] * (1/sqrt(D))
// MODE 1: A=f32 input_kv, store k  f16 [s][h][kv][d] and vT f16 [s][h][d][kv]
// MODE 2: A=f16 attn out, store f32 out[M,N] + bo[n]
static constexpr int BM = 128, BN = 64, BK = 32;
static constexpr int LDA = 40; // padded strides (halves), multiple of 8 -> 16B aligned b128

template <int MODE>
__global__ __launch_bounds__(256) void proj_gemm(const void* __restrict__ Ag,
                                                 const float* __restrict__ Wg,
                                                 const float* __restrict__ bo,
                                                 void* __restrict__ Og, int N) {
  __shared__ _Float16 a_lds[BM * LDA];
  __shared__ _Float16 b_lds[BN * LDA];

  const int t = threadIdx.x;
  const int wave = t >> 5, lane = t & 31;
  const int hb = lane >> 4, nn = lane & 15;
  const int m0 = blockIdx.x * BM;
  const int n0 = blockIdx.y * BN;

  v8f acc[4] = {};

  for (int k0 = 0; k0 < CQ; k0 += BK) {
    __syncthreads();
    // ---- stage A tile: 128 x 32 (convert f32->f16 unless MODE==2) ----
    {
      const int row = t >> 1, cs = (t & 1) * 16;
      if (MODE == 2) {
        const _Float16* A16 = (const _Float16*)Ag;
        const _Float16* src = A16 + (size_t)(m0 + row) * CQ + k0 + cs;
        *(v8h*)(a_lds + row * LDA + cs)     = *(const v8h*)(src);
        *(v8h*)(a_lds + row * LDA + cs + 8) = *(const v8h*)(src + 8);
      } else {
        const float* A32 = (const float*)Ag;
        const float* src = A32 + (size_t)(m0 + row) * CQ + k0 + cs;
        v8h h0, h1;
#pragma unroll
        for (int j = 0; j < 8; ++j) { h0[j] = (_Float16)src[j]; h1[j] = (_Float16)src[8 + j]; }
        *(v8h*)(a_lds + row * LDA + cs)     = h0;
        *(v8h*)(a_lds + row * LDA + cs + 8) = h1;
      }
    }
    // ---- stage B tile transposed: W[k0..k0+32][n0..n0+64] -> b_lds[n][k] ----
    {
      const int kk = t >> 3, ns = (t & 7) * 8;
      const float* src = Wg + (size_t)(k0 + kk) * N + n0 + ns;
#pragma unroll
      for (int j = 0; j < 8; ++j) b_lds[(ns + j) * LDA + kk] = (_Float16)src[j];
    }
    __syncthreads();

    v16h a = load_frag_a(a_lds + wave * 16 * LDA, LDA);
#pragma unroll
    for (int nt = 0; nt < 4; ++nt) {
      v16h b = load_frag_b(b_lds + nt * 16 * LDA, LDA);
      acc[nt] = WMMA_F16(a, b, acc[nt]);
    }
  }

  // ---- epilogue (C layout: lane=(hb,nn); VGPR r -> row hb*8+r, col nn) ----
#pragma unroll
  for (int nt = 0; nt < 4; ++nt) {
#pragma unroll
    for (int r = 0; r < 8; ++r) {
      const int m = hb * 8 + r;
      const int gm = m0 + wave * 16 + m;       // token index
      const int gn = n0 + nt * 16 + nn;        // output column
      const float val = acc[nt][r];
      if (MODE == 0) {
        const int s = gm >> 9, qr = gm & 511;
        const int hh = gn >> 5, d = gn & 31;
        _Float16* qf = (_Float16*)Og;
        qf[(((size_t)s * H_ + hh) * QL + qr) * D_ + d] =
            (_Float16)(val * 0.17677669529663687f); // 1/sqrt(32)
      } else if (MODE == 1) {
        const int s = gm >> 9, kv = gm & 511;
        _Float16* kf  = (_Float16*)Og;
        _Float16* vtf = kf + (size_t)S_ * H_ * KVL * D_;
        if (gn < 256) {
          const int hh = gn >> 5, d = gn & 31;
          kf[(((size_t)s * H_ + hh) * KVL + kv) * D_ + d] = (_Float16)val;
        } else {
          const int c = gn - 256, hh = c >> 5, d = c & 31;
          vtf[(((size_t)s * H_ + hh) * D_ + d) * KVL + kv] = (_Float16)val;
        }
      } else {
        float* outp = (float*)Og;
        outp[(size_t)gm * CQ + gn] = val + bo[gn];
      }
    }
  }
}

// ---------- fused attention: per block = (s, h, 64 q rows), 4 waves ----------
static constexpr int KSTR = 40;   // k_lds row stride (halves)
static constexpr int VSTR = 520;  // vt_lds / p_lds row stride (halves)

__global__ __launch_bounds__(128) void attn_kernel(const _Float16* __restrict__ qf,
                                                   const _Float16* __restrict__ kf,
                                                   const _Float16* __restrict__ vtf,
                                                   const float* __restrict__ mask,
                                                   const float* __restrict__ bias,
                                                   _Float16* __restrict__ of) {
  extern __shared__ char smem_raw[];
  _Float16* k_lds  = (_Float16*)smem_raw;                 // 512*40 halves
  _Float16* vt_lds = k_lds + KVL * KSTR;                  // 32*520
  _Float16* p_lds  = vt_lds + D_ * VSTR;                  // 4*16*520
  float*    scr    = (float*)(p_lds + 4 * 16 * VSTR);     // 4 waves * 48 floats

  const int t = threadIdx.x, wave = t >> 5, lane = t & 31;
  const int hb = lane >> 4, nn = lane & 15;
  const int idx = blockIdx.x;
  const int qblk = idx & 7, h = (idx >> 3) & 7, s = idx >> 6;
  const int qbase = qblk * 64 + wave * 16;

  // ---- stage K [512][32] and V^T [32][512] via CDNA5 async global->LDS ----
  const _Float16* ksrc = kf  + ((size_t)s * H_ + h) * KVL * D_;
  const _Float16* vsrc = vtf + ((size_t)s * H_ + h) * D_ * KVL;
#pragma unroll
  for (int p = 0; p < 16; ++p) {
    const int c = t + p * 128;                 // 2048 chunks of 8 halves
    const int row = c >> 2, seg = c & 3;
    async_copy_b128(ksrc + row * D_ + seg * 8, k_lds + row * KSTR + seg * 8);
  }
#pragma unroll
  for (int p = 0; p < 16; ++p) {
    const int c = t + p * 128;
    const int row = c >> 6, seg = c & 63;
    async_copy_b128(vsrc + row * KVL + seg * 8, vt_lds + row * VSTR + seg * 8);
  }
  async_wait0();           // drain this wave's ASYNCcnt
  __syncthreads();         // make all waves' LDS writes visible

  // ---- q fragment: 16 rows x 32 (scale pre-applied), stays in VGPRs ----
  const _Float16* qsrc = qf + (((size_t)s * H_ + h) * QL + qbase) * D_;
  const v16h qfrag = load_frag_a(qsrc, D_);

  _Float16* p_wave  = p_lds + wave * 16 * VSTR;
  float*    wavescr = scr + wave * 48;

  const float* biasp = bias + ((size_t)h * QL + qbase) * KVL;
  const float* maskp = mask + (size_t)s * KVL;
  __builtin_prefetch(biasp, 0, 1);

  // ---- scores: 32 WMMAs over KV tiles, bias+mask, track row max ----
  float mx[8];
#pragma unroll
  for (int r = 0; r < 8; ++r) mx[r] = -1e30f;

  for (int kt = 0; kt < 32; ++kt) {
    v16h b = load_frag_b(k_lds + kt * 16 * KSTR, KSTR); // B = K^T (K-dim = d)
    v8f c = {};
    c = WMMA_F16(qfrag, b, c);
    const int kv = kt * 16 + nn;
    const float madd = (maskp[kv] - 1.0f) * 1.0e9f;
#pragma unroll
    for (int r = 0; r < 8; ++r) {
      const int m = hb * 8 + r;
      const float sc = c[r] + biasp[(size_t)m * KVL + kv] + madd;
      mx[r] = fmaxf(mx[r], sc);
      p_wave[m * VSTR + kv] = (_Float16)sc;
    }
  }

  // ---- row max across the 16 lanes holding one row ----
#pragma unroll
  for (int r = 0; r < 8; ++r)
#pragma unroll
    for (int off = 1; off < 16; off <<= 1)
      mx[r] = fmaxf(mx[r], __shfl_xor(mx[r], off, 16));
  if (nn < 8) wavescr[hb * 8 + nn] = mx[nn];
  __syncthreads();

  // ---- exp + row sums: lane (hb,nn) handles row nn, half hb of 512 cols ----
  {
    const float rm = wavescr[nn];
    _Float16* prow = p_wave + nn * VSTR + hb * 256;
    float sum = 0.f;
    for (int j = 0; j < 256; j += 8) {
      v8h x = *(v8h*)(prow + j);
#pragma unroll
      for (int e = 0; e < 8; ++e) {
        const float ev = __expf((float)x[e] - rm);
        sum += ev;
        x[e] = (_Float16)ev;
      }
      *(v8h*)(prow + j) = x;
    }
    wavescr[16 + hb * 16 + nn] = sum;
  }
  __syncthreads();
  if (hb == 0) wavescr[nn] = 1.0f / (wavescr[16 + nn] + wavescr[32 + nn]);
  __syncthreads();

  // ---- PV: O[16x32] = P[16x512] @ V[512x32], 32 WMMAs ----
  v8f o0 = {}, o1 = {};
  for (int kt = 0; kt < 16; ++kt) {
    v16h pf = load_frag_a(p_wave + kt * 32, VSTR);
    v16h v0 = load_frag_b(vt_lds + 0 * 16 * VSTR + kt * 32, VSTR);
    v16h v1 = load_frag_b(vt_lds + 16 * VSTR + kt * 32, VSTR);
    o0 = WMMA_F16(pf, v0, o0);
    o1 = WMMA_F16(pf, v1, o1);
  }

  // ---- epilogue: fold 1/sum, store f16 [token][h*32+d] for the O-proj ----
#pragma unroll
  for (int r = 0; r < 8; ++r) {
    const int m = hb * 8 + r;
    const float inv = wavescr[m];
    const size_t token = (size_t)s * QL + qbase + m;
    of[token * (H_ * D_) + h * D_ + nn]      = (_Float16)(o0[r] * inv);
    of[token * (H_ * D_) + h * D_ + 16 + nn] = (_Float16)(o1[r] * inv);
  }
}

// ---------- host launcher ----------
extern "C" void kernel_launch(void* const* d_in, const int* in_sizes, int n_in,
                              void* d_out, int out_size, void* d_ws, size_t ws_size,
                              hipStream_t stream) {
  (void)in_sizes; (void)n_in; (void)out_size; (void)ws_size;
  const float* input_q  = (const float*)d_in[0];
  const float* input_kv = (const float*)d_in[1];
  const float* mask     = (const float*)d_in[2];
  const float* bias     = (const float*)d_in[3];
  const float* Wq       = (const float*)d_in[4];
  const float* Wkv      = (const float*)d_in[5];
  const float* Wo       = (const float*)d_in[6];
  const float* bo       = (const float*)d_in[7];
  float* out = (float*)d_out;

  const size_t perTensor = (size_t)S_ * H_ * QL * D_; // 8.39M halves each
  _Float16* qf  = (_Float16*)d_ws;
  _Float16* kf  = qf + perTensor;
  _Float16* vtf = kf + perTensor;
  _Float16* of  = vtf + perTensor;

  const int Mblocks = (S_ * QL) / BM; // 256

  proj_gemm<0><<<dim3(Mblocks, CQ / BN), dim3(256), 0, stream>>>(input_q, Wq, bo, qf, 256);
  proj_gemm<1><<<dim3(Mblocks, 512 / BN), dim3(256), 0, stream>>>(input_kv, Wkv, bo, kf, 512);

  const size_t attn_lds =
      (size_t)(KVL * KSTR + D_ * VSTR + 4 * 16 * VSTR) * sizeof(_Float16) +
      4 * 48 * sizeof(float); // ~138 KB (within 320 KB WGP LDS)
  attn_kernel<<<dim3(S_ * H_ * 8), dim3(128), attn_lds, stream>>>(qf, kf, vtf, mask, bias, of);

  proj_gemm<2><<<dim3(Mblocks, CQ / BN), dim3(256), 0, stream>>>(of, Wo, bo, out, 256);
}
